// PINTO_2D_1408749273194
// MI455X (gfx1250) — matrix operand order repaired and checked by reading, not compiled, not run on hardware
//
#include <hip/hip_runtime.h>
#include <math.h>

typedef _Float16 h8   __attribute__((ext_vector_type(8)));
typedef _Float16 v16h __attribute__((ext_vector_type(16)));
typedef float    v8f  __attribute__((ext_vector_type(8)));

#define PADH 136   // halves per f16 tile row (272B stride, 16B-aligned sublanes)
#define PADF 132   // floats per f32 residual tile row (528B stride)
#define DM   128
#define HD   32
#define NH   4
#define NL   3

__device__ __forceinline__ v8f splat8(float x) {
    v8f v;
#pragma unroll
    for (int i = 0; i < 8; ++i) v[i] = x;
    return v;
}

__device__ __forceinline__ v8f wmma_f16(v16h a, v16h b, v8f c) {
    return __builtin_amdgcn_wmma_f32_16x16x32_f16(false, a, false, b, (short)0, c, false, false);
}

// A-fragment (16x32 f16, M=ln, K-halves per ISA table) from per-wave LDS tile
__device__ __forceinline__ v16h load_afrag(const _Float16* tile, int ln, int hiHalf, int kc) {
    int base = kc * 32 + (hiHalf ? 8 : 0);
    h8 lo = *(const h8*)(tile + ln * PADH + base);
    h8 hi = *(const h8*)(tile + ln * PADH + base + 16);
    return __builtin_shufflevector(lo, hi, 0,1,2,3,4,5,6,7,8,9,10,11,12,13,14,15);
}

// B-fragment (32x16 f16): lane -> N column (lane%16), elements -> K per ISA table (lane half selects K-lo/hi)
__device__ __forceinline__ v16h load_bfrag(const _Float16* w, int ldw, int kc, int nc, int lane) {
    int hh = lane >> 4, n = nc * 16 + (lane & 15);
    v16h b;
#pragma unroll
    for (int s = 0; s < 16; ++s) {
        int k = (s < 8) ? (8 * hh + s) : (16 + 8 * hh + (s - 8));
        b[s] = w[(size_t)(kc * 32 + k) * ldw + n];
    }
    return b;
}

__device__ __forceinline__ float gelu_exact(float x) {
    return 0.5f * x * (1.0f + erff(x * 0.70710678118654752f));
}

__device__ __forceinline__ void feats30(float x0, float x1, float* f) {
    const float PIc = 3.14159265358979323846f;
#pragma unroll
    for (int j = 0; j < 2; ++j) {
        float x = j == 0 ? x0 : x1;
        f[0*2+j]  = x;
        f[1*2+j]  = x * x;
        f[2*2+j]  = (x - 1.5f) * (x - 1.5f);
        f[3*2+j]  = (x - 4.5f) * (x - 4.5f);
        f[4*2+j]  = 1.f / (1.f + __expf(-x));
        f[5*2+j]  = 1.f / (1.f + __expf(-(x - 1.5f)));
        f[6*2+j]  = 1.f / (1.f + __expf(-(x - 4.5f)));
        f[7*2+j]  = x - 1.5f;
        f[8*2+j]  = x - 4.5f;
        f[9*2+j]  = __sinf(PIc * x);
        f[10*2+j] = __cosf(PIc * x);
        f[11*2+j] = __sinf(PIc * x * 0.25f);
        f[12*2+j] = __cosf(PIc * x * 0.25f);
        f[13*2+j] = __sinf(PIc * x * 0.5f);
        f[14*2+j] = __cosf(PIc * x * 0.5f);
    }
}

// ---------------- prep: repack/convert weights to f16, zero reduction buffers ----------------
__global__ void prep_kernel(const float* __restrict__ Wq, const float* __restrict__ Wo,
                            const float* __restrict__ W1, const float* __restrict__ W2,
                            const float* __restrict__ pw1, const float* __restrict__ pe_w,
                            _Float16* wq16, _Float16* wo16, _Float16* w116, _Float16* w216,
                            _Float16* pw116, _Float16* pew16, float* kvf32, float* ksf32) {
    int i = blockIdx.x * 256 + threadIdx.x;
    const int WSZ = NL * DM * DM;
    if (i < WSZ) {
        int l = i / (DM * DM);
        int rem = i - l * DM * DM;
        int d = rem >> 7;
        int n = rem & 127;
        int h = n >> 5, e = n & 31;
        wq16[i] = (_Float16)Wq[(((size_t)l * NH + h) * DM + d) * HD + e]; // head-major cols
        wo16[i] = (_Float16)Wo[i];
        w116[i] = (_Float16)W1[i];
        w216[i] = (_Float16)W2[i];
    }
    if (i < DM * 64) pw116[i] = (_Float16)pw1[i];
    if (i < 32 * DM) {                       // pe_w padded 30 -> 32 rows
        int row = i >> 7;
        pew16[i] = (row < 30) ? (_Float16)pe_w[i] : (_Float16)0.f;
    }
    if (i < NL * NH * HD * HD) kvf32[i] = 0.f;
    if (i < NL * NH * HD) ksf32[i] = 0.f;
}

// ---------------- boundary encodings ----------------
__global__ void benc_kernel(const float* __restrict__ bpts, const float* __restrict__ bpe_w,
                            const float* __restrict__ bpe_b, float* __restrict__ benc, int M) {
    int i = blockIdx.x * 256 + threadIdx.x;
    if (i >= M * DM) return;
    int m = i >> 7, d = i & 127;
    float f[30];
    feats30(bpts[m * 2], bpts[m * 2 + 1], f);
    float a = bpe_b[d];
#pragma unroll
    for (int ff = 0; ff < 30; ++ff) a += f[ff] * bpe_w[ff * DM + d];
    benc[i] = __sinf(a);
}

// ---------------- kv_sum / k_sum reduction over boundary points ----------------
// grid = (L*H) * 32 blocks; block 256 threads = 8 groups of 32 lanes
__global__ void kv_kernel(const float* __restrict__ benc,
                          const float* __restrict__ Wk, const float* __restrict__ bk,
                          const float* __restrict__ Wv, const float* __restrict__ bv,
                          float* kvf32, float* ksf32, int M) {
    int lh = blockIdx.x >> 5, chunk = blockIdx.x & 31;
    int g = threadIdx.x >> 5, e = threadIdx.x & 31;
    const float* wk = Wk + (size_t)lh * DM * HD;
    const float* wv = Wv + (size_t)lh * DM * HD;
    float bkk = bk[lh * HD + e], bvv = bv[lh * HD + e];
    float kvc[32];
#pragma unroll
    for (int d = 0; d < 32; ++d) kvc[d] = 0.f;
    float ks = 0.f;
    int rows = M / 32;
    int m0 = chunk * rows;
    for (int m = m0 + g; m < m0 + rows; m += 8) {
        const float* br = benc + (size_t)m * DM;
        float kh = bkk, vh = bvv;
#pragma unroll 8
        for (int d = 0; d < DM; ++d) {
            float b = br[d];
            kh += b * wk[d * HD + e];
            vh += b * wv[d * HD + e];
        }
        kh = kh * kh;   // squared feature map
        ks += kh;
#pragma unroll
        for (int d = 0; d < 32; ++d) kvc[d] += __shfl(kh, d, 32) * vh;
    }
    __shared__ float red[8][32][32];
#pragma unroll
    for (int d = 0; d < 32; ++d) red[g][d][e] = kvc[d];
    __shared__ float red2[8][32];
    red2[g][e] = ks;
    __syncthreads();
    int t = threadIdx.x;
    for (int idx = t; idx < 1024; idx += 256) {
        float s = 0.f;
#pragma unroll
        for (int g2 = 0; g2 < 8; ++g2) s += red[g2][idx >> 5][idx & 31];
        atomicAdd(&kvf32[lh * 1024 + idx], s);
    }
    if (t < 32) {
        float s = 0.f;
#pragma unroll
        for (int g2 = 0; g2 < 8; ++g2) s += red2[g2][t];
        atomicAdd(&ksf32[lh * HD + t], s);
    }
}

__global__ void kvfin_kernel(const float* kvf32, _Float16* kv16) {
    int i = blockIdx.x * 256 + threadIdx.x;
    if (i < NL * NH * HD * HD) kv16[i] = (_Float16)kvf32[i];
}

// ---------------- fused main kernel: 16 query rows per wave, residual tile in LDS ----------------
__global__ __launch_bounds__(128, 1)
void fused_main(const float* __restrict__ qpts, const _Float16* __restrict__ pew16,
                const float* __restrict__ pe_b,
                const _Float16* __restrict__ wq16, const float* __restrict__ bq,
                const _Float16* __restrict__ kv16, const float* __restrict__ ksum,
                const _Float16* __restrict__ wo16, const float* __restrict__ bo,
                const _Float16* __restrict__ w116, const float* __restrict__ b1,
                const _Float16* __restrict__ w216, const float* __restrict__ b2,
                const _Float16* __restrict__ pw116, const float* __restrict__ pb1,
                const float* __restrict__ pw2, const float* __restrict__ pb2,
                float* __restrict__ out) {
    __shared__ __align__(32) _Float16 smemA[4][16 * PADH];  // matmul staging (f16)
    __shared__ __align__(16) float    smemR[4][16 * PADF];  // residual h/x (f32)
    int wid = threadIdx.x >> 5;
    int lane = threadIdx.x & 31;
    int ln = lane & 15;
    int hiHalf = lane >> 4;          // 0: rows 0-7 / K-lo, 1: rows 8-15 / K-hi
    int mb = hiHalf ? 8 : 0;
    _Float16* tile = smemA[wid];
    float*    tR   = smemR[wid];
    int row0 = (blockIdx.x * 4 + wid) * 16;

    // ---- positional encoding via WMMA: h = sin(feat30pad32 @ pe_w16 + pe_b) ----
    {
        // cooperative feats tile: row = ln; lo lane fills cols 0-14 (+30,31 zero), hi lane 15-29
        float f[30];
        feats30(qpts[(size_t)(row0 + ln) * 2], qpts[(size_t)(row0 + ln) * 2 + 1], f);
        _Float16* trow = tile + ln * PADH;
        if (!hiHalf) {
#pragma unroll
            for (int ff = 0; ff < 15; ++ff) trow[ff] = (_Float16)f[ff];
            trow[30] = (_Float16)0.f;
            trow[31] = (_Float16)0.f;
        } else {
#pragma unroll
            for (int ff = 15; ff < 30; ++ff) trow[ff] = (_Float16)f[ff];
        }
        v16h a0 = load_afrag(tile, ln, hiHalf, 0);
        // h -> tR (f32); f16 copy written to tile AFTER all frag consumers (next loop reads tile? no:
        // a0 is in registers, so overwriting tile below is safe)
#pragma unroll 2
        for (int nc = 0; nc < 8; ++nc) {
            v8f acc = splat8(pe_b[nc * 16 + ln]);
            acc = wmma_f16(a0, load_bfrag(pew16, DM, 0, nc, lane), acc);
            int n = nc * 16 + ln;
#pragma unroll
            for (int r = 0; r < 8; ++r) {
                float hv = __sinf(acc[r]);
                tR[(r + mb) * PADF + n] = hv;
                tile[(r + mb) * PADH + n] = (_Float16)hv;
            }
        }
    }

    for (int l = 0; l < NL; ++l) {
        const _Float16* Wql = wq16 + (size_t)l * DM * DM;
        const _Float16* Wol = wo16 + (size_t)l * DM * DM;
        const _Float16* W1l = w116 + (size_t)l * DM * DM;
        const _Float16* W2l = w216 + (size_t)l * DM * DM;
        const float* bql = bq + l * DM;
        const float* bol = bo + l * DM;
        const float* b1l = b1 + l * DM;
        const float* b2l = b2 + l * DM;
        const _Float16* KVl = kv16 + (size_t)l * NH * HD * HD;
        const float* KSl = ksum + l * NH * HD;

        v16h af[4];
#pragma unroll
        for (int kc = 0; kc < 4; ++kc) af[kc] = load_afrag(tile, ln, hiHalf, kc);

        // ---- q = (h @ Wq + bq)^2, store q^2 to tile, accumulate denominator partials ----
        // FULLY unrolled: d4/z4 must keep compile-time indices (register arrays)
        float d4[4][8];
#pragma unroll
        for (int h = 0; h < 4; ++h)
#pragma unroll
            for (int r = 0; r < 8; ++r) d4[h][r] = 0.f;
#pragma unroll
        for (int nc = 0; nc < 8; ++nc) {
            v8f acc = splat8(bql[nc * 16 + ln]);
#pragma unroll
            for (int kc = 0; kc < 4; ++kc)
                acc = wmma_f16(af[kc], load_bfrag(Wql, DM, kc, nc, lane), acc);
            float kss = KSl[nc * 16 + ln];
            int n = nc * 16 + ln;
#pragma unroll
            for (int r = 0; r < 8; ++r) {
                float t = acc[r];
                t = t * t;
                tile[(r + mb) * PADH + n] = (_Float16)t;
                d4[nc >> 1][r] += t * kss;
            }
        }
        // z = 1/(q2 . k_sum + 1e-6), half-wave reduction (rows match C-layout)
        float z4[4][8];
#pragma unroll
        for (int h = 0; h < 4; ++h) {
#pragma unroll
            for (int r = 0; r < 8; ++r) {
                float dd = d4[h][r];
                dd += __shfl_xor(dd, 1, 32);
                dd += __shfl_xor(dd, 2, 32);
                dd += __shfl_xor(dd, 4, 32);
                dd += __shfl_xor(dd, 8, 32);
                z4[h][r] = 1.f / (dd + 1e-6f);
            }
        }
        // ---- attn = (q2 @ kv_sum) * z per head (fully unrolled: z4[h] compile-time) ----
#pragma unroll
        for (int h = 0; h < 4; ++h) {
            v16h afh = load_afrag(tile, ln, hiHalf, h);
#pragma unroll
            for (int j = 0; j < 2; ++j) {
                v16h bf = load_bfrag(KVl + (size_t)h * 1024 + j * 16, 32, 0, 0, lane);
                v8f acc = wmma_f16(afh, bf, splat8(0.f));
                int n = (2 * h + j) * 16 + ln;
#pragma unroll
                for (int r = 0; r < 8; ++r)
                    tile[(r + mb) * PADH + n] = (_Float16)(acc[r] * z4[h][r]);
            }
        }
        // ---- x = h + attn @ Wo + bo : residual read/write through tR ----
#pragma unroll
        for (int kc = 0; kc < 4; ++kc) af[kc] = load_afrag(tile, ln, hiHalf, kc);
#pragma unroll 2
        for (int nc = 0; nc < 8; ++nc) {
            v8f acc = splat8(bol[nc * 16 + ln]);
#pragma unroll
            for (int kc = 0; kc < 4; ++kc)
                acc = wmma_f16(af[kc], load_bfrag(Wol, DM, kc, nc, lane), acc);
            int n = nc * 16 + ln;
#pragma unroll
            for (int r = 0; r < 8; ++r) {
                float xv = tR[(r + mb) * PADF + n] + acc[r];
                tR[(r + mb) * PADF + n] = xv;
                tile[(r + mb) * PADH + n] = (_Float16)xv;
            }
        }
        // ---- FFN: u = gelu(x @ W1 + b1) ----
#pragma unroll
        for (int kc = 0; kc < 4; ++kc) af[kc] = load_afrag(tile, ln, hiHalf, kc);
#pragma unroll 2
        for (int nc = 0; nc < 8; ++nc) {
            v8f acc = splat8(b1l[nc * 16 + ln]);
#pragma unroll
            for (int kc = 0; kc < 4; ++kc)
                acc = wmma_f16(af[kc], load_bfrag(W1l, DM, kc, nc, lane), acc);
            int n = nc * 16 + ln;
#pragma unroll
            for (int r = 0; r < 8; ++r)
                tile[(r + mb) * PADH + n] = (_Float16)gelu_exact(acc[r]);
        }
        // ---- h' = x + u @ W2 + b2 ----
#pragma unroll
        for (int kc = 0; kc < 4; ++kc) af[kc] = load_afrag(tile, ln, hiHalf, kc);
#pragma unroll 2
        for (int nc = 0; nc < 8; ++nc) {
            v8f acc = splat8(b2l[nc * 16 + ln]);
#pragma unroll
            for (int kc = 0; kc < 4; ++kc)
                acc = wmma_f16(af[kc], load_bfrag(W2l, DM, kc, nc, lane), acc);
            int n = nc * 16 + ln;
#pragma unroll
            for (int r = 0; r < 8; ++r) {
                float hv = tR[(r + mb) * PADF + n] + acc[r];
                tR[(r + mb) * PADF + n] = hv;
                tile[(r + mb) * PADH + n] = (_Float16)hv;
            }
        }
    }

    // ---- final projector: gelu(h @ pw1 + pb1) @ pw2 + pb2 ----
    {
        v16h af[4];
#pragma unroll
        for (int kc = 0; kc < 4; ++kc) af[kc] = load_afrag(tile, ln, hiHalf, kc);
#pragma unroll 2
        for (int nc = 0; nc < 4; ++nc) {
            v8f acc = splat8(pb1[nc * 16 + ln]);
#pragma unroll
            for (int kc = 0; kc < 4; ++kc)
                acc = wmma_f16(af[kc], load_bfrag(pw116, 64, kc, nc, lane), acc);
            int n = nc * 16 + ln;
#pragma unroll
            for (int r = 0; r < 8; ++r)
                tR[(r + mb) * PADF + n] = gelu_exact(acc[r]);
        }
    }
    if (lane < 16) {   // per-row 64->3 scalar projection from tR
        int m = lane;
        float s0 = pb2[0], s1 = pb2[1], s2 = pb2[2];
#pragma unroll 8
        for (int k = 0; k < 64; ++k) {
            float p = tR[m * PADF + k];
            s0 += p * pw2[k * 3 + 0];
            s1 += p * pw2[k * 3 + 1];
            s2 += p * pw2[k * 3 + 2];
        }
        size_t row = (size_t)(row0 + m);
        out[row * 3 + 0] = s0;
        out[row * 3 + 1] = s1;
        out[row * 3 + 2] = s2;
    }
}

extern "C" void kernel_launch(void* const* d_in, const int* in_sizes, int n_in,
                              void* d_out, int out_size, void* d_ws, size_t ws_size,
                              hipStream_t stream) {
    const float* qpts  = (const float*)d_in[0];
    const float* bpts  = (const float*)d_in[1];
    const float* pe_w  = (const float*)d_in[2];
    const float* pe_b  = (const float*)d_in[3];
    const float* bpe_w = (const float*)d_in[4];
    const float* bpe_b = (const float*)d_in[5];
    const float* Wq    = (const float*)d_in[6];
    const float* bq    = (const float*)d_in[7];
    const float* Wk    = (const float*)d_in[8];
    const float* bk    = (const float*)d_in[9];
    const float* Wv    = (const float*)d_in[10];
    const float* bv    = (const float*)d_in[11];
    const float* Wo    = (const float*)d_in[12];
    const float* bo    = (const float*)d_in[13];
    const float* W1    = (const float*)d_in[14];
    const float* b1    = (const float*)d_in[15];
    const float* W2    = (const float*)d_in[16];
    const float* b2    = (const float*)d_in[17];
    const float* pw1   = (const float*)d_in[18];
    const float* pb1   = (const float*)d_in[19];
    const float* pw2   = (const float*)d_in[20];
    const float* pb2   = (const float*)d_in[21];
    float* out = (float*)d_out;

    int N = in_sizes[0] / 2;
    int M = in_sizes[1] / 2;

    char* ws = (char*)d_ws;
    size_t off = 0;
    float* benc = (float*)(ws + off);        off += (size_t)M * DM * sizeof(float);
    _Float16* wq16 = (_Float16*)(ws + off);  off += (size_t)NL * DM * DM * 2;
    _Float16* wo16 = (_Float16*)(ws + off);  off += (size_t)NL * DM * DM * 2;
    _Float16* w116 = (_Float16*)(ws + off);  off += (size_t)NL * DM * DM * 2;
    _Float16* w216 = (_Float16*)(ws + off);  off += (size_t)NL * DM * DM * 2;
    _Float16* pw116 = (_Float16*)(ws + off); off += (size_t)DM * 64 * 2;
    _Float16* pew16 = (_Float16*)(ws + off); off += (size_t)32 * DM * 2;
    _Float16* kv16 = (_Float16*)(ws + off);  off += (size_t)NL * NH * HD * HD * 2;
    float* kvf32 = (float*)(ws + off);       off += (size_t)NL * NH * HD * HD * sizeof(float);
    float* ksf32 = (float*)(ws + off);       off += (size_t)NL * NH * HD * sizeof(float);

    prep_kernel<<<(NL * DM * DM + 255) / 256, 256, 0, stream>>>(
        Wq, Wo, W1, W2, pw1, pe_w, wq16, wo16, w116, w216, pw116, pew16, kvf32, ksf32);
    benc_kernel<<<(M * DM + 255) / 256, 256, 0, stream>>>(bpts, bpe_w, bpe_b, benc, M);
    kv_kernel<<<NL * NH * 32, 256, 0, stream>>>(benc, Wk, bk, Wv, bv, kvf32, ksf32, M);
    kvfin_kernel<<<(NL * NH * HD * HD + 255) / 256, 256, 0, stream>>>(kvf32, kv16);
    fused_main<<<N / 64, 128, 0, stream>>>(
        qpts, pew16, pe_b, wq16, bq, kv16, ksf32, wo16, bo,
        w116, b1, w216, b2, pw116, pb1, pw2, pb2, out);
}
